// MultiHead_SelfAttention_9981503996413
// MI455X (gfx1250) — compile-verified, matrix operand order checked by reading, not run on hardware
//
#include <hip/hip_runtime.h>
#include <hip/hip_bf16.h>

// ---------------------------------------------------------------------------
// MultiHead self-attention with softmax over the HEAD axis (per reference).
// Pass 1: qkv = x @ W^T + b  (bf16 WMMA GEMM), stores bf16 Q[b,h,n,d],
//         K[b,h,n,d], and V transposed Vt[b,h,d,n] into d_ws (48 MB).
// Pass 2: fused streaming attention. One WG = one (b, 16-row n-tile);
//         16 waves, wave h owns head h. For each 32-col m-tile:
//           s = q @ k^T (WMMA), e = exp(s) -> LDS E[pos][head],
//           thread t owns softmax position t: z = sum_h E[t][h] (4x b128),
//           P[h][t] = e * rcp(z) in bf16,
//           O += P @ V (WMMA, Vt gives contiguous-K B-frags).
// ---------------------------------------------------------------------------

typedef __attribute__((ext_vector_type(16))) __bf16 v16bf;
typedef __attribute__((ext_vector_type(2)))  __bf16 v2bf;
typedef __attribute__((ext_vector_type(2)))  float  v2f;
typedef __attribute__((ext_vector_type(8)))  float  v8f;

union FragBF {
    v16bf    v;
    unsigned u[8];
};

__device__ inline unsigned short f2bf(float f) {
    union { __bf16 b; unsigned short s; } u;
    u.b = (__bf16)f;                       // backend RNE conversion
    return u.s;
}

__device__ inline unsigned f2bf_pair(float lo, float hi) {
    v2f f = { lo, hi };
    union { v2bf b; unsigned u; } u;
    u.b = __builtin_convertvector(f, v2bf); // v_cvt_pk_bf16_f32 class
    return u.u;
}

// ---------------------------------------------------------------------------
// Pass 1: QKV projection GEMM.  M = B*N = 8192, Nc = 3*C = 3072, K = 1024.
// Grid: 64 row-tiles (128 rows) x 48 col-tiles (64 cols) = 3072 WGs, 256 thr.
// Wave w computes rows [w*16, w*16+16) x 64 cols (4 16x16 f32 accumulators).
// W tile (64 cols x 32 k) staged through double-buffered LDS as bf16.
// ---------------------------------------------------------------------------
__global__ __launch_bounds__(256) void qkv_gemm_kernel(
    const float* __restrict__ x, const float* __restrict__ W,
    const float* __restrict__ bias,
    unsigned short* __restrict__ Qb, unsigned short* __restrict__ Kb,
    unsigned short* __restrict__ Vt)
{
    __shared__ unsigned Ws[2][64 * 16];   // double buffer: 64 n x 32 k bf16 pairs

    const int tid  = threadIdx.x;
    const int lane = tid & 31;
    const int wave = tid >> 5;
    const int l16  = lane & 15;
    const int hiH  = lane >> 4;

    const int r0 = (blockIdx.x / 48) * 128;
    const int j0 = (blockIdx.x % 48) * 64;
    const int i0 = r0 + wave * 16;

    // K-pair offsets for 16-bit A/B WMMA fragments (ISA 7.12.2)
    int kOff[8];
#pragma unroll
    for (int v = 0; v < 8; ++v)
        kOff[v] = ((v < 4) ? 2 * v : 8 + 2 * v) + 8 * hiH;

    v8f acc[4] = {};

    for (int k0 = 0, it = 0; k0 < 1024; k0 += 32, ++it) {
        unsigned* wbuf = &Ws[it & 1][0];
        // cooperative stage of W tile (f32 -> bf16)
#pragma unroll
        for (int q = 0; q < 4; ++q) {
            int p  = tid + 256 * q;          // dword-pair index 0..1023
            int n  = p >> 4;
            int kk = (p & 15) * 2;
            const float2 w2 =
                *reinterpret_cast<const float2*>(&W[(j0 + n) * 1024 + k0 + kk]);
            wbuf[p] = f2bf_pair(w2.x, w2.y);
        }
        __syncthreads();

        // A fragment straight from global x (f32 -> bf16)
        FragBF a;
        const float* xrow = &x[(size_t)(i0 + l16) * 1024 + k0];
#pragma unroll
        for (int v = 0; v < 8; ++v) {
            const float2 x2 = *reinterpret_cast<const float2*>(&xrow[kOff[v]]);
            a.u[v] = f2bf_pair(x2.x, x2.y);
        }

#pragma unroll
        for (int jt = 0; jt < 4; ++jt) {
            FragBF bf;
            const int N = jt * 16 + l16;
#pragma unroll
            for (int v = 0; v < 8; ++v)
                bf.u[v] = wbuf[N * 16 + (kOff[v] >> 1)];
            acc[jt] = __builtin_amdgcn_wmma_f32_16x16x32_bf16(
                false, a.v, false, bf.v, (short)0, acc[jt], false, false);
        }
        // no second barrier: alternate buffer is safe by program order
    }

    // epilogue: bias add + scatter into Q / K / V^T bf16 tensors
#pragma unroll
    for (int jt = 0; jt < 4; ++jt) {
        const int j     = j0 + jt * 16 + l16;
        const float bv  = bias[j];
        const int three = j >> 10;          // 0=q 1=k 2=v (uniform per jt-tile)
        const int h     = (j >> 6) & 15;
        const int d     = j & 63;
#pragma unroll
        for (int r = 0; r < 8; ++r) {
            const int i  = i0 + r + 8 * hiH;
            const int b  = i >> 11;
            const int n  = i & 2047;
            const int bh = b * 16 + h;
            const unsigned short val = f2bf(acc[jt][r] + bv);
            if (three == 0)      Qb[(bh * 2048 + n) * 64 + d]  = val;
            else if (three == 1) Kb[(bh * 2048 + n) * 64 + d]  = val;
            else                 Vt[(bh * 64 + d) * 2048 + n]  = val;  // transposed
        }
    }
}

// ---------------------------------------------------------------------------
// Pass 2: fused attention with head-axis softmax.
// Grid: B * N/16 = 512 WGs, 512 threads (16 waves, wave = head).
// ---------------------------------------------------------------------------
__global__ __launch_bounds__(512) void attn_head_softmax_kernel(
    const unsigned short* __restrict__ Qb,
    const unsigned short* __restrict__ Kb,
    const unsigned short* __restrict__ Vt,
    float* __restrict__ out)
{
    __shared__ float          E[512][16];   // exp(s): [softmax position][head]
    __shared__ unsigned short P[16][512];   // normalized probs bf16, per head

    const int tid  = threadIdx.x;
    const int lane = tid & 31;
    const int h    = tid >> 5;              // head = wave
    const int l16  = lane & 15;
    const int hiH  = lane >> 4;

    const int b  = blockIdx.x >> 7;         // 128 n-tiles per batch
    const int n0 = (blockIdx.x & 127) * 16;
    const int bh = b * 16 + h;

    int kOff[8];
#pragma unroll
    for (int v = 0; v < 8; ++v)
        kOff[v] = ((v < 4) ? 2 * v : 8 + 2 * v) + 8 * hiH;

    // Q A-fragments (16 rows x 64 d), held for the whole kernel
    FragBF qf[2];
    const unsigned short* qrow = &Qb[(bh * 2048 + n0 + l16) * 64];
#pragma unroll
    for (int kd = 0; kd < 2; ++kd)
#pragma unroll
        for (int v = 0; v < 8; ++v)
            qf[kd].u[v] =
                *reinterpret_cast<const unsigned*>(&qrow[kd * 32 + kOff[v]]);

    const unsigned short* Kbase = &Kb[bh * 2048 * 64];
    const unsigned short* Vbase = &Vt[bh * 64 * 2048];

    v8f o[4] = {};

    for (int m0 = 0; m0 < 2048; m0 += 32) {
        // prefetch next K/V tiles (global_prefetch_b8)
        if (m0 + 32 < 2048) {
            __builtin_prefetch(&Kbase[(m0 + 32 + l16) * 64], 0, 1);
            __builtin_prefetch(&Vbase[l16 * 2048 + m0 + 32], 0, 1);
        }

        // ---- scores: s[16 x 32] = q @ k^T --------------------------------
        v8f s[2] = {};
#pragma unroll
        for (int mt = 0; mt < 2; ++mt) {
            const int m = m0 + mt * 16 + l16;     // B-frag column = k row
#pragma unroll
            for (int kd = 0; kd < 2; ++kd) {
                FragBF kf;
#pragma unroll
                for (int v = 0; v < 8; ++v)
                    kf.u[v] = *reinterpret_cast<const unsigned*>(
                        &Kbase[m * 64 + kd * 32 + kOff[v]]);
                s[mt] = __builtin_amdgcn_wmma_f32_16x16x32_bf16(
                    false, qf[kd].v, false, kf.v, (short)0, s[mt], false, false);
            }
        }

        // ---- e = exp(s) scattered to E[pos][head] ------------------------
#pragma unroll
        for (int mt = 0; mt < 2; ++mt)
#pragma unroll
            for (int r = 0; r < 8; ++r)
                E[(r + 8 * hiH) * 32 + mt * 16 + l16][h] = __expf(s[mt][r]);
        __syncthreads();

        // ---- thread tid owns softmax position tid ------------------------
        {
            float ev[16];
            *reinterpret_cast<float4*>(&ev[0])  = *reinterpret_cast<const float4*>(&E[tid][0]);
            *reinterpret_cast<float4*>(&ev[4])  = *reinterpret_cast<const float4*>(&E[tid][4]);
            *reinterpret_cast<float4*>(&ev[8])  = *reinterpret_cast<const float4*>(&E[tid][8]);
            *reinterpret_cast<float4*>(&ev[12]) = *reinterpret_cast<const float4*>(&E[tid][12]);
            float z = 0.f;
#pragma unroll
            for (int hh = 0; hh < 16; ++hh) z += ev[hh];   // fixed order: deterministic
            const float rz = __builtin_amdgcn_rcpf(z);
#pragma unroll
            for (int hh = 0; hh < 16; ++hh)
                P[hh][tid] = f2bf(ev[hh] * rz);
        }
        __syncthreads();

        // ---- P A-frag (bf16, C->A layout via LDS) ------------------------
        FragBF pf;
#pragma unroll
        for (int v = 0; v < 8; ++v)
            pf.u[v] = *reinterpret_cast<const unsigned*>(&P[h][l16 * 32 + kOff[v]]);

        // ---- O[16 x 64] += P[16 x 32] @ V[32 x 64]  (Vt: contiguous K) ---
#pragma unroll
        for (int dt = 0; dt < 4; ++dt) {
            FragBF vf;
#pragma unroll
            for (int v = 0; v < 8; ++v)
                vf.u[v] = *reinterpret_cast<const unsigned*>(
                    &Vbase[(dt * 16 + l16) * 2048 + m0 + kOff[v]]);
            o[dt] = __builtin_amdgcn_wmma_f32_16x16x32_bf16(
                false, pf.v, false, vf.v, (short)0, o[dt], false, false);
        }
        // E overwrite next iter is after sync#2; P overwrite after next sync#1:
        // both ordered w.r.t. this iteration's readers by program order.
    }

    // ---- write out[b, n, h*64 + d] (f32) ---------------------------------
#pragma unroll
    for (int dt = 0; dt < 4; ++dt)
#pragma unroll
        for (int r = 0; r < 8; ++r) {
            const int n = n0 + r + 8 * hiH;
            const int c = h * 64 + dt * 16 + l16;
            out[(size_t)(b * 2048 + n) * 1024 + c] = o[dt][r];
        }
}

// ---------------------------------------------------------------------------
extern "C" void kernel_launch(void* const* d_in, const int* in_sizes, int n_in,
                              void* d_out, int out_size, void* d_ws, size_t ws_size,
                              hipStream_t stream)
{
    (void)in_sizes; (void)n_in; (void)out_size; (void)ws_size;

    const float* x    = (const float*)d_in[0];   // (4, 2048, 1024)
    const float* W    = (const float*)d_in[1];   // (3072, 1024)
    const float* bias = (const float*)d_in[2];   // (3072,)
    float* out        = (float*)d_out;           // (4, 2048, 1024)

    const size_t elems = (size_t)4 * 16 * 2048 * 64;   // 8,388,608 per tensor
    unsigned short* Qb = (unsigned short*)d_ws;
    unsigned short* Kb = Qb + elems;
    unsigned short* Vt = Kb + elems;                   // 48 MB total scratch

    qkv_gemm_kernel<<<dim3(3072), dim3(256), 0, stream>>>(x, W, bias, Qb, Kb, Vt);
    attn_head_softmax_kernel<<<dim3(512), dim3(512), 0, stream>>>(Qb, Kb, Vt, out);
}
